// LSTMActor_36747740184872
// MI455X (gfx1250) — matrix operand order repaired and checked by reading, compiled, not run on hardware
//
#include <hip/hip_runtime.h>
#include <hip/hip_bf16.h>

// ---------------------------------------------------------------------------
// Fused 2-layer LSTM + FC actor for MI455X (gfx1250, wave32, WMMA 16x16x32 f16)
//
// One wave owns a 16-row batch tile. Per timestep:
//   layer1: 2x v_wmma_f32_16x16x32_f16  (K = [x(2), h1(8)] zero-padded to 32)
//   layer2: 4x v_wmma                   (K = [h1(8), h2(16)] zero-padded to 32)
//   every 2 steps: 8x v_wmma accumulate FC1 (K = [h2_t, h2_{t+1}])
// Activations are branch-free single-tanh forms (v_tanh_f32 when available).
// ---------------------------------------------------------------------------

typedef _Float16 half_t;
typedef __attribute__((ext_vector_type(16))) _Float16 v16h;
typedef __attribute__((ext_vector_type(8)))  float    v8f;

#define T_STEPS 96
#define WAVES   2      // waves per block
#define ROWS    16     // batch rows per wave (WMMA M)
#define OBS_W   194
#define OBS_PAD 200

union FragU { uint4 u[2]; v16h h; };
union PackU { half_t h[2]; unsigned u; };

__device__ __forceinline__ v8f wmma16(v16h a, v16h b, v8f c) {
  // D = A(16x32 f16) * B(32x16 f16) + C(16x16 f32)
  return __builtin_amdgcn_wmma_f32_16x16x32_f16(false, a, false, b, (short)0, c,
                                                false, false);
}

// A-fragment (16x32 f16) from LDS rows of 32 halves each, stored in K-order.
// Layout: lanes 0-15 (row M=lane) hold K=0..15
//         lanes 16-31 (row M=lane-16) hold K=16..31
__device__ __forceinline__ v16h load_a_frag(const half_t* base, int lane) {
  const uint4* p = (const uint4*)(base + (lane & 15) * 32 + ((lane & 16) ? 16 : 0));
  FragU f;
  f.u[0] = p[0];   // 8 halves: K = kbase .. kbase+7
  f.u[1] = p[1];   // 8 halves: K = kbase+8 .. kbase+15
  return f.h;
}

// B-fragment (32x16 f16): pre-swizzled in workspace, 32B per lane, contiguous.
__device__ __forceinline__ v16h load_b_frag(const uint4* g, int lane) {
  FragU f;
  f.u[0] = g[lane * 2];
  f.u[1] = g[lane * 2 + 1];
  return f.h;
}

// Branch-free tanh: prefer gfx1250 hardware v_tanh_f32.
__device__ __forceinline__ float fast_tanh(float x) {
#if __has_builtin(__builtin_amdgcn_tanhf)
  return __builtin_amdgcn_tanhf(x);
#elif __has_builtin(__builtin_amdgcn_tanh_f32)
  return __builtin_amdgcn_tanh_f32(x);
#else
  // tanh(x) = 1 - 2/(1+exp(2x)); saturates correctly at +/-inf, no branches.
  float e = __expf(2.0f * x);
  return 1.0f - 2.0f * __builtin_amdgcn_rcpf(1.0f + e);
#endif
}

// sigmoid(x) = 0.5 + 0.5*tanh(0.5x) : single trans op, branch-free.
__device__ __forceinline__ float fast_sigm(float x) {
  return __builtin_fmaf(0.5f, fast_tanh(0.5f * x), 0.5f);
}

// ---------------------------------------------------------------------------
// Prep kernel: swizzle all weights into WMMA B-fragment layout (f16) in ws.
//   Per tile (32 K x 16 N): lane n    holds N=n, K=0..15  (16 halves)
//                           lane n+16 holds N=n, K=16..31
// ---------------------------------------------------------------------------
__global__ void prep_kernel(const float* __restrict__ W_ih1,
                            const float* __restrict__ W_hh1,
                            const float* __restrict__ b_ih1,
                            const float* __restrict__ b_hh1,
                            const float* __restrict__ W_ih2,
                            const float* __restrict__ W_hh2,
                            const float* __restrict__ b_ih2,
                            const float* __restrict__ b_hh2,
                            const float* __restrict__ W_fc1,
                            half_t* __restrict__ wsB1,   // 2 tiles * 512 halves
                            half_t* __restrict__ wsB2,   // 4 tiles * 512 halves
                            float*  __restrict__ bias1,  // 32
                            float*  __restrict__ bias2,  // 64
                            half_t* __restrict__ wsFC)   // 49*8 tiles * 512 halves
{
  const int tid = blockIdx.x * blockDim.x + threadIdx.x;
  const int nth = gridDim.x * blockDim.x;

  // Layer1 B: K = [x0, x1, h1[0..7], 0-pad]; N = gate 0..31
  for (int i = tid; i < 2 * 512; i += nth) {
    int tile = i >> 9, lane = (i >> 4) & 31, w = i & 15;
    int k = ((lane & 16) ? 16 : 0) + w;
    int n = tile * 16 + (lane & 15);
    float v = 0.0f;
    if (k < 2)       v = W_ih1[n * 2 + k];
    else if (k < 10) v = W_hh1[n * 8 + (k - 2)];
    wsB1[i] = (half_t)v;
  }
  // Layer2 B: K = [h1[0..7], h2[0..15], 0-pad]; N = gate 0..63
  for (int i = tid; i < 4 * 512; i += nth) {
    int tile = i >> 9, lane = (i >> 4) & 31, w = i & 15;
    int k = ((lane & 16) ? 16 : 0) + w;
    int n = tile * 16 + (lane & 15);
    float v = 0.0f;
    if (k < 8)       v = W_ih2[n * 8 + k];
    else if (k < 24) v = W_hh2[n * 16 + (k - 8)];
    wsB2[i] = (half_t)v;
  }
  // FC1 B: 48 K-tiles of [h2_t(16), h2_{t+1}(16)] + 1 tile for UNIQ x2.
  // W_fc1 is (128, 1538); concat column for (t,h) is t*16+h; x2 -> 1536,1537.
  for (int i = tid; i < 49 * 8 * 512; i += nth) {
    int tile = i >> 9;            // kt*8 + nt
    int kt = tile >> 3, nt = tile & 7;
    int lane = (i >> 4) & 31, w = i & 15;
    int koff = ((lane & 16) ? 16 : 0) + w;
    int n = nt * 16 + (lane & 15);
    float v = 0.0f;
    if (kt < 48) {
      int t = kt * 2 + (koff >> 4);
      int h = koff & 15;
      v = W_fc1[n * 1538 + t * 16 + h];
    } else {
      if (koff == 0)      v = W_fc1[n * 1538 + 1536];
      else if (koff == 1) v = W_fc1[n * 1538 + 1537];
    }
    wsFC[i] = (half_t)v;
  }
  // Combined gate biases
  for (int i = tid; i < 32; i += nth) bias1[i] = b_ih1[i] + b_hh1[i];
  for (int i = tid; i < 64; i += nth) bias2[i] = b_ih2[i] + b_hh2[i];
}

// ---------------------------------------------------------------------------
// Main fused kernel
// ---------------------------------------------------------------------------
__global__ __launch_bounds__(WAVES * 32) void lstm_actor_kernel(
    const float* __restrict__ obs,
    const float* __restrict__ b_fc1,
    const float* __restrict__ W_fc2,
    const float* __restrict__ b_fc2,
    const uint4* __restrict__ wsB1,   // 2 tiles * 64 uint4
    const uint4* __restrict__ wsB2,   // 4 tiles
    const float* __restrict__ bias1,
    const float* __restrict__ bias2,
    const uint4* __restrict__ wsFC,   // 49*8 tiles
    float* __restrict__ out)
{
  __shared__ half_t obsL  [WAVES][ROWS][OBS_PAD]; // f16 copy of obs rows
  __shared__ half_t a1row [WAVES][ROWS][32];      // layer1 A rows (K-order)
  __shared__ half_t a2row [WAVES][ROWS][32];      // layer2 A rows
  __shared__ half_t h2pair[WAVES][ROWS][32];      // [h2_t | h2_{t+1}] for FC1
  __shared__ float  gates1[WAVES][32][ROWS];      // [N][M]
  __shared__ float  gates2[WAVES][64][ROWS];
  __shared__ float  fcbuf [WAVES][128][ROWS];     // relu(FC1) [N][M]
  __shared__ float  wfc2L [2][128];

  const int wv   = threadIdx.x >> 5;
  const int lane = threadIdx.x & 31;
  const int ncol = lane & 15;               // this lane's N column in a tile
  const int m0   = (lane & 16) ? 8 : 0;     // C-fragment row offset
  const int bbase = blockIdx.x * (WAVES * ROWS) + wv * ROWS;

  // ---- init: obs -> LDS (f16), W_fc2 -> LDS, zero A rows (h=0 initial) ----
  for (int r = 0; r < ROWS; ++r) {
    const float* src = obs + (size_t)(bbase + r) * OBS_W;
    for (int c = lane; c < OBS_W; c += 32)
      obsL[wv][r][c] = (half_t)src[c];
  }
  for (int k = lane; k < ROWS * 32; k += 32) {
    a1row[wv][k >> 5][k & 31] = (half_t)0.0f;
    a2row[wv][k >> 5][k & 31] = (half_t)0.0f;
  }
  for (int k = threadIdx.x; k < 256; k += WAVES * 32)
    wfc2L[k >> 7][k & 127] = W_fc2[k];
  __syncthreads();

  // ---- persistent B fragments (weights) ----
  v16h B1f0 = load_b_frag(wsB1, lane);
  v16h B1f1 = load_b_frag(wsB1 + 64, lane);
  v16h B2f[4];
#pragma unroll
  for (int i = 0; i < 4; ++i) B2f[i] = load_b_frag(wsB2 + i * 64, lane);

  // per-lane bias values (column ncol of each tile)
  const float bg1_0 = bias1[ncol];
  const float bg1_1 = bias1[16 + ncol];
  float bg2[4];
#pragma unroll
  for (int i = 0; i < 4; ++i) bg2[i] = bias2[i * 16 + ncol];

  // Layer1 tile1 activation: N=16..23 -> tanh(g), N=24..31 -> sigmoid(o).
  // Branch-free: y = ofs + amp * tanh(scl * x), per-lane constants.
  const bool  isg    = (ncol < 8);
  const float g1_scl = isg ? 1.0f : 0.5f;
  const float g1_amp = isg ? 1.0f : 0.5f;
  const float g1_ofs = isg ? 0.0f : 0.5f;

  // ---- FC1 accumulators, init with b_fc1 ----
  v8f accFC[8];
#pragma unroll
  for (int nt = 0; nt < 8; ++nt) {
    float b = b_fc1[nt * 16 + ncol];
#pragma unroll
    for (int v = 0; v < 8; ++v) accFC[nt][v] = b;
  }

  // ---- cell states in registers of the updating lanes ----
  float c1s[4] = {0.f, 0.f, 0.f, 0.f};
  float c2s[8] = {0.f, 0.f, 0.f, 0.f, 0.f, 0.f, 0.f, 0.f};

  // One LSTM timestep (both layers); phase selects h2pair half.
  auto lstm_step = [&](int t, int phase) {
    // x_t = [obs[b,t], obs[b,96+t]] into K=0,1 of layer1 A rows.
    // Lanes l and l+16 write the same dword (same data) -> no divergence.
    {
      PackU px;
      px.h[0] = obsL[wv][ncol][t];
      px.h[1] = obsL[wv][ncol][96 + t];
      *(unsigned*)&a1row[wv][ncol][0] = px.u;
    }

    // ---------------- layer 1 ----------------
    {
      v16h a1 = load_a_frag(&a1row[wv][0][0], lane);
      v8f g0, g1;
#pragma unroll
      for (int v = 0; v < 8; ++v) { g0[v] = bg1_0; g1[v] = bg1_1; }
      g0 = wmma16(a1, B1f0, g0);   // gates i(0..7), f(8..15)
      g1 = wmma16(a1, B1f1, g1);   // gates g(16..23), o(24..31)
#pragma unroll
      for (int v = 0; v < 8; ++v) {
        g0[v] = fast_sigm(g0[v]);
        g1[v] = __builtin_fmaf(g1_amp, fast_tanh(g1_scl * g1[v]), g1_ofs);
      }
#pragma unroll
      for (int v = 0; v < 8; ++v) {
        gates1[wv][ncol][m0 + v]      = g0[v];
        gates1[wv][16 + ncol][m0 + v] = g1[v];
      }
      // update 4 (row, h) elements per lane: c1 in regs, h1 -> LDS (f16)
#pragma unroll
      for (int j = 0; j < 4; ++j) {
        int idx = lane * 4 + j;
        int r = idx >> 3, h = idx & 7;
        float iv = gates1[wv][h][r];
        float fv = gates1[wv][8 + h][r];
        float gv = gates1[wv][16 + h][r];
        float ov = gates1[wv][24 + h][r];
        float c  = fv * c1s[j] + iv * gv;
        c1s[j] = c;
        half_t hh = (half_t)(ov * fast_tanh(c));
        a1row[wv][r][2 + h] = hh;   // next step's layer1 input
        a2row[wv][r][h]     = hh;   // this step's layer2 input
      }
    }

    // ---------------- layer 2 ----------------
    {
      v16h a2 = load_a_frag(&a2row[wv][0][0], lane);
      v8f q[4];
#pragma unroll
      for (int nt = 0; nt < 4; ++nt) {
#pragma unroll
        for (int v = 0; v < 8; ++v) q[nt][v] = bg2[nt];
        q[nt] = wmma16(a2, B2f[nt], q[nt]);
      }
      // i(0..15) sig, f(16..31) sig, g(32..47) tanh, o(48..63) sig
#pragma unroll
      for (int v = 0; v < 8; ++v) {
        q[0][v] = fast_sigm(q[0][v]);
        q[1][v] = fast_sigm(q[1][v]);
        q[2][v] = fast_tanh(q[2][v]);
        q[3][v] = fast_sigm(q[3][v]);
      }
#pragma unroll
      for (int nt = 0; nt < 4; ++nt)
#pragma unroll
        for (int v = 0; v < 8; ++v)
          gates2[wv][nt * 16 + ncol][m0 + v] = q[nt][v];
      // update 8 (row, h) elements per lane
#pragma unroll
      for (int j = 0; j < 8; ++j) {
        int idx = lane * 8 + j;
        int r = idx >> 4, h = idx & 15;
        float iv = gates2[wv][h][r];
        float fv = gates2[wv][16 + h][r];
        float gv = gates2[wv][32 + h][r];
        float ov = gates2[wv][48 + h][r];
        float c  = fv * c2s[j] + iv * gv;
        c2s[j] = c;
        half_t hh = (half_t)(ov * fast_tanh(c));
        a2row[wv][r][8 + h]          = hh;  // next step's h2 input
        h2pair[wv][r][phase * 16 + h] = hh; // FC1 K-tile buffer
      }
    }
  };

  // =================== time loop: 48 double-steps ===================
  for (int kt = 0; kt < 48; ++kt) {
    // Hoist this iteration's FC1 weight fragments (L2-resident) so their
    // latency overlaps the two LSTM steps; prefetch the next K-tile.
    const uint4* base = wsFC + (size_t)kt * (8 * 64);
    v16h bf[8];
#pragma unroll
    for (int nt = 0; nt < 8; ++nt) bf[nt] = load_b_frag(base + nt * 64, lane);
    __builtin_prefetch(wsFC + (size_t)(kt + 1) * (8 * 64) + lane * 2, 0, 1);

    lstm_step(2 * kt, 0);
    lstm_step(2 * kt + 1, 1);

    v16h af = load_a_frag(&h2pair[wv][0][0], lane);
#pragma unroll
    for (int nt = 0; nt < 8; ++nt)
      accFC[nt] = wmma16(af, bf[nt], accFC[nt]);
  }

  // ---- final K-tile: UNIQ features x2 (K=0,1 only) ----
  {
    FragU ax;
    ax.u[0] = make_uint4(0u, 0u, 0u, 0u);
    ax.u[1] = make_uint4(0u, 0u, 0u, 0u);
    if (lane < 16) {
      ax.h[0] = obsL[wv][lane][192];
      ax.h[1] = obsL[wv][lane][193];
    }
    const uint4* base = wsFC + (size_t)48 * (8 * 64);
#pragma unroll
    for (int nt = 0; nt < 8; ++nt) {
      v16h bf = load_b_frag(base + nt * 64, lane);
      accFC[nt] = wmma16(ax.h, bf, accFC[nt]);
    }
  }

  // ---- ReLU, spill FC1 to LDS ----
#pragma unroll
  for (int nt = 0; nt < 8; ++nt)
#pragma unroll
    for (int v = 0; v < 8; ++v)
      fcbuf[wv][nt * 16 + ncol][m0 + v] = fmaxf(accFC[nt][v], 0.0f);

  // ---- FC2 (128 -> 2) + softsign; lane -> (row = lane&15, o = lane>>4) ----
  {
    int r = lane & 15, o = lane >> 4;
    float acc = b_fc2[o];
#pragma unroll 4
    for (int k = 0; k < 128; ++k)
      acc += fcbuf[wv][k][r] * wfc2L[o][k];
    float y = acc * __builtin_amdgcn_rcpf(1.0f + fabsf(acc));
    out[(size_t)(bbase + r) * 2 + o] = y;
  }
}

// ---------------------------------------------------------------------------
// Launch
// ---------------------------------------------------------------------------
extern "C" void kernel_launch(void* const* d_in, const int* in_sizes, int n_in,
                              void* d_out, int out_size, void* d_ws, size_t ws_size,
                              hipStream_t stream) {
  const float* obs   = (const float*)d_in[0];
  const float* W_ih1 = (const float*)d_in[1];
  const float* W_hh1 = (const float*)d_in[2];
  const float* b_ih1 = (const float*)d_in[3];
  const float* b_hh1 = (const float*)d_in[4];
  const float* W_ih2 = (const float*)d_in[5];
  const float* W_hh2 = (const float*)d_in[6];
  const float* b_ih2 = (const float*)d_in[7];
  const float* b_hh2 = (const float*)d_in[8];
  const float* W_fc1 = (const float*)d_in[9];
  const float* b_fc1 = (const float*)d_in[10];
  const float* W_fc2 = (const float*)d_in[11];
  const float* b_fc2 = (const float*)d_in[12];
  float* out = (float*)d_out;

  // Workspace layout (bytes): B1 frags @0 (2 KB), B2 @2048 (4 KB),
  // bias1 @6144, bias2 @6272, FC1 frags @8192 (49*8*1024 = 392 KB).
  char* ws = (char*)d_ws;
  half_t* wsB1  = (half_t*)(ws + 0);
  half_t* wsB2  = (half_t*)(ws + 2048);
  float*  bias1 = (float*) (ws + 6144);
  float*  bias2 = (float*) (ws + 6272);
  half_t* wsFC  = (half_t*)(ws + 8192);

  prep_kernel<<<512, 256, 0, stream>>>(W_ih1, W_hh1, b_ih1, b_hh1,
                                       W_ih2, W_hh2, b_ih2, b_hh2,
                                       W_fc1, wsB1, wsB2, bias1, bias2, wsFC);

  const int batch = 65536;
  const int rows_per_block = WAVES * ROWS;             // 32
  lstm_actor_kernel<<<batch / rows_per_block, WAVES * 32, 0, stream>>>(
      obs, b_fc1, W_fc2, b_fc2,
      (const uint4*)wsB1, (const uint4*)wsB2, bias1, bias2,
      (const uint4*)wsFC, out);
}
